// GraphSAGEFraudDetector_80522046865741
// MI455X (gfx1250) — matrix operand order repaired and checked by reading, not compiled, not run on hardware
//
#include <hip/hip_runtime.h>
#include <math.h>

typedef __attribute__((ext_vector_type(16))) _Float16 v16h;
typedef __attribute__((ext_vector_type(8)))  _Float16 h8;
typedef __attribute__((ext_vector_type(8)))  float    v8f;

#define GEMM_WAVES 4   // waves per block
#define GEMM_NT    4   // 16-wide N-tiles per wave (A-fragment reuse)

// ---------------- utility kernels ----------------

__global__ void zero_f32(float* __restrict__ p, long long n) {
    long long i = (long long)blockIdx.x * blockDim.x + threadIdx.x;
    long long stride = (long long)gridDim.x * blockDim.x;
    for (; i < n; i += stride) p[i] = 0.0f;
}

__global__ void cvt_f16(const float* __restrict__ in, _Float16* __restrict__ out,
                        long long n) {
    long long i = (long long)blockIdx.x * blockDim.x + threadIdx.x;
    long long stride = (long long)gridDim.x * blockDim.x;
    for (; i < n; i += stride) out[i] = (_Float16)in[i];
}

__global__ void deg_count(const int* __restrict__ dst, float* __restrict__ deg, int E) {
    int e = blockIdx.x * blockDim.x + threadIdx.x;
    if (e < E) atomicAdd(&deg[dst[e]], 1.0f);
}

__global__ void invert_deg(float* __restrict__ deg, int N) {
    int i = blockIdx.x * blockDim.x + threadIdx.x;
    if (i < N) deg[i] = 1.0f / fmaxf(deg[i], 1.0f);
}

// one block per edge; lanes stride over features (coalesced atomics)
__global__ void scatter_add(const float* __restrict__ h, const int* __restrict__ src,
                            const int* __restrict__ dst, float* __restrict__ agg, int F) {
    int e = blockIdx.x;
    int s = src[e];
    int d = dst[e];
    const float* hrow = h + (long long)s * F;
    float*       arow = agg + (long long)d * F;
    for (int f = threadIdx.x; f < F; f += blockDim.x)
        atomicAdd(&arow[f], hrow[f]);
}

// mean-scale the fp32 accumulation, emit f16 GEMM operand
__global__ void scale_rows_f16(const float* __restrict__ agg, const float* __restrict__ inv,
                               int F, long long total, _Float16* __restrict__ out) {
    long long i = (long long)blockIdx.x * blockDim.x + threadIdx.x;
    if (i < total) out[i] = (_Float16)(agg[i] * inv[i / F]);   // F pow2 -> shift
}

// ---------------- WMMA GEMM (f16 operands, f32 accumulate) ----------------
// out[Nrows, Hout] = act( concat(A0[Nrows,F0], A1[Nrows,F1]) @ W[Hout,K]^T + bias )
// All of A0/A1/W are pre-converted f16; B[k][n] = W[n*K + k].

__device__ inline h8 load8h_concat(const _Float16* __restrict__ A0,
                                   const _Float16* __restrict__ A1,
                                   int F0, int F1, long long row, int k) {
    // 8 contiguous K values; F0,F1 multiples of 32 so the segment is in one half
    const _Float16* p = (k < F0) ? (A0 + row * F0 + k)
                                 : (A1 + row * F1 + (k - F0));
    return *(const h8*)p;   // 16B load
}

__global__ __launch_bounds__(32 * GEMM_WAVES)
void wmma_gemm_concat(const _Float16* __restrict__ A0, int F0,
                      const _Float16* __restrict__ A1, int F1,
                      const _Float16* __restrict__ W,
                      const float* __restrict__ bias,
                      float* __restrict__ out,
                      _Float16* __restrict__ out_h,   // optional f16 mirror (may be null)
                      int Nrows, int Hout, int do_relu) {
    const int K    = F0 + F1;
    const int lane = threadIdx.x & 31;
    const int wave = threadIdx.x >> 5;
    const int lrow = lane & 15;   // row/col within 16x16 tile
    const int hi   = lane >> 4;   // half-wave select

    const int m_base = blockIdx.x * 16;
    const int n_base = (blockIdx.y * GEMM_WAVES + wave) * (16 * GEMM_NT);
    if (m_base >= Nrows) return;

    v8f acc[GEMM_NT];
    for (int t = 0; t < GEMM_NT; ++t)
        for (int r = 0; r < 8; ++r) acc[t][r] = 0.0f;

    long long arow = m_base + lrow;
    if (arow >= Nrows) arow = Nrows - 1;   // clamp (N divisible by 16 here)

    for (int k0 = 0; k0 < K; k0 += 32) {
        // A fragment, 16x32 f16 ISA layout:
        //  lanes 0-15 : e=0..7 -> K=k0+e,  e=8..15 -> K=k0+16+(e-8)
        //  lanes 16-31: same +8
        h8 a_lo = load8h_concat(A0, A1, F0, F1, arow, k0 + hi * 8);
        h8 a_hi = load8h_concat(A0, A1, F0, F1, arow, k0 + 16 + hi * 8);
        v16h afrag;
        for (int j = 0; j < 8; ++j) { afrag[j] = a_lo[j]; afrag[8 + j] = a_hi[j]; }

        // B fragments (32x16 f16): lane holds column n, e=0..7 -> K=k0+hi*8+e, e=8..15 -> K=k0+16+hi*8+(e-8)
        for (int t = 0; t < GEMM_NT; ++t) {
            const int n = n_base + t * 16 + lrow;
            const _Float16* p = W + (long long)n * K + k0 + hi * 8;
            h8 b_lo = *(const h8*)p;
            h8 b_hi = *(const h8*)(p + 16);
            v16h bfrag;
            for (int j = 0; j < 8; ++j) { bfrag[j] = b_lo[j]; bfrag[8 + j] = b_hi[j]; }

            acc[t] = __builtin_amdgcn_wmma_f32_16x16x32_f16(
                false, afrag, false, bfrag, (short)0, acc[t], false, false);
        }
    }

    // epilogue: C/D layout VGPR r -> M = m_base + r + 8*hi, N = lane%16
    for (int t = 0; t < GEMM_NT; ++t) {
        const int n = n_base + t * 16 + lrow;
        const float b = bias[n];
        for (int r = 0; r < 8; ++r) {
            const int m = m_base + r + 8 * hi;
            if (m < Nrows) {
                float v = acc[t][r] + b;
                if (do_relu) v = fmaxf(v, 0.0f);
                out[(long long)m * Hout + n] = v;
                if (out_h) out_h[(long long)m * Hout + n] = (_Float16)v;
            }
        }
    }
}

// ---------------- sigmoid head: one wave per node ----------------

__global__ void head_kernel(const float* __restrict__ h2, const float* __restrict__ Wo,
                            const float* __restrict__ bo, float* __restrict__ out,
                            int N, int H) {
    const int lane = threadIdx.x & 31;
    const int wave = threadIdx.x >> 5;
    const int node = blockIdx.x * (blockDim.x >> 5) + wave;
    if (node >= N) return;
    const float* row = h2 + (long long)node * H;
    float s = 0.0f;
    for (int i = lane; i < H; i += 32) s += row[i] * Wo[i];
    for (int off = 16; off > 0; off >>= 1) s += __shfl_xor(s, off, 32);
    if (lane == 0) {
        float z = s + bo[0];
        out[node] = 1.0f / (1.0f + __expf(-z));
    }
}

// ---------------- launch ----------------

static inline size_t align256(size_t x) { return (x + 255) & ~(size_t)255; }

extern "C" void kernel_launch(void* const* d_in, const int* in_sizes, int n_in,
                              void* d_out, int out_size, void* d_ws, size_t ws_size,
                              hipStream_t stream) {
    const float* x    = (const float*)d_in[0];
    const int*   edge = (const int*)  d_in[1];
    const float* W1   = (const float*)d_in[2];
    const float* b1   = (const float*)d_in[3];
    const float* W2   = (const float*)d_in[4];
    const float* b2   = (const float*)d_in[5];
    const float* Wo   = (const float*)d_in[6];
    const float* bo   = (const float*)d_in[7];

    const int F = 128, H = 512;
    const int N = in_sizes[0] / F;      // 50000
    const int E = in_sizes[1] / 2;      // 400000
    const int* src = edge;              // edge_index[0]
    const int* dst = edge + E;          // edge_index[1]

    // ---- workspace bump allocator (256B aligned) ----
    char* base = (char*)d_ws;
    size_t off = 0;
    auto alloc = [&](size_t bytes) { char* p = base + off; off += align256(bytes); return p; };

    float*    inv   = (float*)   alloc((size_t)N * sizeof(float));            // deg -> 1/deg
    float*    agg   = (float*)   alloc((size_t)N * H * sizeof(float));        // fp32 scatter scratch (both layers)
    _Float16* xh    = (_Float16*)alloc((size_t)N * F * sizeof(_Float16));     // x in f16
    _Float16* aggh  = (_Float16*)alloc((size_t)N * H * sizeof(_Float16));     // agg1h / agg2h (not live together)
    float*    h1    = (float*)   alloc((size_t)N * H * sizeof(float));        // layer-1 act (fp32, for scatter)
    _Float16* h1h   = (_Float16*)alloc((size_t)N * H * sizeof(_Float16));     // layer-1 act (f16, GEMM2 A)
    _Float16* W1h   = (_Float16*)alloc((size_t)H * 2 * F * sizeof(_Float16));
    _Float16* W2h   = (_Float16*)alloc((size_t)H * 2 * H * sizeof(_Float16));
    float*    h2    = agg;   // agg fp32 is dead once agg2h exists -> reuse for h2

    // ---- degrees -> inv_deg ----
    zero_f32<<<1024, 256, 0, stream>>>(inv, (long long)N);
    deg_count<<<(E + 255) / 256, 256, 0, stream>>>(dst, inv, E);
    invert_deg<<<(N + 255) / 256, 256, 0, stream>>>(inv, N);

    // ---- one-time f16 materialization of GEMM operands ----
    cvt_f16<<<2048, 256, 0, stream>>>(x,  xh,  (long long)N * F);
    cvt_f16<<<256,  256, 0, stream>>>(W1, W1h, (long long)H * 2 * F);
    cvt_f16<<<1024, 256, 0, stream>>>(W2, W2h, (long long)H * 2 * H);

    // ---- layer-1 mean aggregation (fp32 accumulate, f16 out) ----
    const long long a1n = (long long)N * F;
    zero_f32<<<4096, 256, 0, stream>>>(agg, a1n);
    scatter_add<<<E, 128, 0, stream>>>(x, src, dst, agg, F);
    scale_rows_f16<<<(int)((a1n + 255) / 256), 256, 0, stream>>>(agg, inv, F, a1n, aggh);

    // ---- GEMM1: h1 = relu([x | n1] @ W1^T + b1), dual fp32+f16 output ----
    dim3 g1((N + 15) / 16, H / (16 * GEMM_NT * GEMM_WAVES));   // (3125, 2)
    wmma_gemm_concat<<<g1, 32 * GEMM_WAVES, 0, stream>>>(
        xh, F, aggh, F, W1h, b1, h1, h1h, N, H, 1);

    // ---- layer-2 mean aggregation ----
    const long long a2n = (long long)N * H;
    zero_f32<<<8192, 256, 0, stream>>>(agg, a2n);
    scatter_add<<<E, 256, 0, stream>>>(h1, src, dst, agg, H);
    scale_rows_f16<<<(int)((a2n + 255) / 256), 256, 0, stream>>>(agg, inv, H, a2n, aggh);

    // ---- GEMM2: h2 = relu([h1 | n2] @ W2^T + b2) (h2 aliases agg) ----
    wmma_gemm_concat<<<g1, 32 * GEMM_WAVES, 0, stream>>>(
        h1h, H, aggh, H, W2h, b2, h2, (_Float16*)nullptr, N, H, 1);

    // ---- sigmoid head ----
    head_kernel<<<(N + 7) / 8, 256, 0, stream>>>(h2, Wo, bo, (float*)d_out, N, H);
}
